// DNCLM_32804960207199
// MI455X (gfx1250) — compile-verified, matrix operand order checked by reading, not run on hardware
//
#include <hip/hip_runtime.h>
#include <math.h>

// ---------------------------------------------------------------------------
// DNC-LM forward for gfx1250 (MI455X).
//   R=2, W=16, M=500, NHID=512, NINP=512, T=35, B=20, IFACE=93, NTOKEN=32000
// GEMMs use V_WMMA_F32_16X16X32_F16 (f16 in, f32 accumulate), wave32.
// Fragment loads are vectorized: all K segments are multiples of 32, every
// 8-element K-run sits in one segment -> two aligned float4 loads per run.
// ---------------------------------------------------------------------------

typedef __attribute__((ext_vector_type(16))) _Float16 v16h;
typedef __attribute__((ext_vector_type(8)))  float    v8f;

#define T_LEN 35
#define BATCH 20
#define NHID  512
#define NINP  512
#define MCELLS 500
#define WCELL 16
#define RHEADS 2
#define NTOKEN 32000
#define XDIM 544          /* NINP + R*W (layer-0 input incl. zero last_read) */
#define YDIM 544          /* NHID + R*W */
#define DELTA 1e-6f
#define CLIPV 20.0f

// ---------------- device helpers -------------------------------------------

__device__ __forceinline__ float sigf(float x) { return 1.0f / (1.0f + __expf(-x)); }
__device__ __forceinline__ float softplusf_(float x) { return (x > 20.0f) ? x : log1pf(__expf(x)); }

// Pack 8 consecutive f32 (16-byte aligned) into halfs [base, base+8) of v.
__device__ __forceinline__ void pack8(v16h& v, int base, const float* __restrict__ src) {
    const float4 x0 = *(const float4*)(src);
    const float4 x1 = *(const float4*)(src + 4);
    v[base + 0] = (_Float16)x0.x;
    v[base + 1] = (_Float16)x0.y;
    v[base + 2] = (_Float16)x0.z;
    v[base + 3] = (_Float16)x0.w;
    v[base + 4] = (_Float16)x1.x;
    v[base + 5] = (_Float16)x1.y;
    v[base + 6] = (_Float16)x1.z;
    v[base + 7] = (_Float16)x1.w;
}

// A fragment: 16x32 f16, per-lane: row = m0+(lane&15), K runs at
// kc = k0+(lane>>4)*8: halfs 0..7 -> K=kc..kc+7, halfs 8..15 -> K=kc+16..kc+23.
// Two-segment A: k<K0 from A0 (lda0), else from A1 (lda1). Rows >= M read 0.
// REQUIRES: K0 % 32 == 0, lda % 4 == 0 (holds for every call site).
__device__ __forceinline__ v16h load_a_frag(const float* __restrict__ A0, int lda0, int K0,
                                            const float* __restrict__ A1, int lda1, int K1,
                                            int row, int M, int kc) {
    v16h v = {};
    if (row >= M) return v;
#pragma unroll
    for (int c = 0; c < 2; ++c) {
        int k = kc + c * 16;
        const float* src = (k < K0) ? (A0 + (size_t)row * lda0 + k)
                                    : (A1 + (size_t)row * lda1 + (k - K0));
        pack8(v, c * 8, src);
    }
    return v;
}

// B fragment: 32x16 f16 where B[k][n] = W[n][k] (weights row-major N x K),
// two segments W0 (NxK0) / W1 (NxK1). Rows n >= NW read 0.
__device__ __forceinline__ v16h load_b_frag(const float* __restrict__ W0,
                                            const float* __restrict__ W1,
                                            int K0, int K1, int n, int NW, int kc) {
    v16h v = {};
    if (n >= NW) return v;
#pragma unroll
    for (int c = 0; c < 2; ++c) {
        int k = kc + c * 16;
        const float* src = (k < K0) ? (W0 + (size_t)n * K0 + k)
                                    : (W1 + (size_t)n * K1 + (k - K0));
        pack8(v, c * 8, src);
    }
    return v;
}

// ---------------- generic WMMA GEMM -----------------------------------------
// C[M x N] = [A0|A1] (M x (K0+K1), f32) @ [W0|W1]^T + bias0 + bias1
// grid: x = ceil(N/64) (4 waves/block, 16 cols per wave), y = ceil(M/32).
__global__ void __launch_bounds__(128)
wmma_gemm2(const float* __restrict__ A0, int lda0, int K0,
           const float* __restrict__ A1, int lda1, int K1,
           const float* __restrict__ W0, const float* __restrict__ W1, int NW,
           const float* __restrict__ b0, const float* __restrict__ b1, int biasN,
           float* __restrict__ C, int ldc, int M, int N) {
    int wave = threadIdx.x >> 5;
    int lane = threadIdx.x & 31;
    int n0 = (blockIdx.x * 4 + wave) * 16;
    if (n0 >= N) return;
    int m0 = blockIdx.y * 32;
    int K  = K0 + K1;

    v8f acc0 = {};
    v8f acc1 = {};

    int mA   = m0 + (lane & 15);
    int koff = (lane >> 4) * 8;
    int nB   = n0 + (lane & 15);

    for (int kb = 0; kb < K; kb += 32) {
        v16h a0 = load_a_frag(A0, lda0, K0, A1, lda1, K1, mA,      M, kb + koff);
        v16h a1 = load_a_frag(A0, lda0, K0, A1, lda1, K1, mA + 16, M, kb + koff);
        v16h bf = load_b_frag(W0, W1, K0, K1, nB, NW, kb + koff);
        acc0 = __builtin_amdgcn_wmma_f32_16x16x32_f16(false, a0, false, bf,
                                                      (short)0, acc0, false, false);
        acc1 = __builtin_amdgcn_wmma_f32_16x16x32_f16(false, a1, false, bf,
                                                      (short)0, acc1, false, false);
    }

    // C/D layout: VGPR i: lanes 0-15 -> M=i, lanes 16-31 -> M=8+i, N=lane&15.
    int col = n0 + (lane & 15);
    int rb  = (lane < 16) ? 0 : 8;
    float bias = 0.0f;
    if (col < biasN) {
        if (b0) bias += b0[col];
        if (b1) bias += b1[col];
    }
#pragma unroll
    for (int i = 0; i < 8; ++i) {
        int r0 = m0 + rb + i;
        if (r0 < M) C[(size_t)r0 * ldc + col] = acc0[i] + bias;
        int r1 = m0 + 16 + rb + i;
        if (r1 < M) C[(size_t)r1 * ldc + col] = acc1[i] + bias;
    }
}

// ---------------- small kernels ---------------------------------------------

__global__ void fill_kernel(float* p, float v, int n) {
    int i = blockIdx.x * blockDim.x + threadIdx.x;
    if (i < n) p[i] = v;
}

__global__ void embed_kernel(const int* __restrict__ tok,
                             const float* __restrict__ emb_w,
                             float* __restrict__ xbuf) {
    int idx = blockIdx.x * blockDim.x + threadIdx.x;
    int total = T_LEN * BATCH * XDIM;
    if (idx >= total) return;
    int j  = idx % XDIM;
    int tb = idx / XDIM;
    // ixaxiom quirk: last 32 columns (initial last_read) are always zero.
    xbuf[idx] = (j < NINP) ? emb_w[(size_t)tok[tb] * NINP + j] : 0.0f;
}

__global__ void lstm_pw_kernel(const float* __restrict__ gates, int ldg,
                               float* __restrict__ h, float* __restrict__ c,
                               float* __restrict__ outc) {
    int idx = blockIdx.x * blockDim.x + threadIdx.x;
    if (idx >= BATCH * NHID) return;
    int b = idx / NHID, j = idx % NHID;
    const float* g = gates + (size_t)b * ldg;
    float gi = g[j], gf = g[NHID + j], gg = g[2 * NHID + j], go = g[3 * NHID + j];
    float cn = sigf(gf) * c[idx] + sigf(gi) * tanhf(gg);
    c[idx] = cn;
    float hn = sigf(go) * tanhf(cn);
    h[idx] = hn;
    if (outc) outc[idx] = fminf(CLIPV, fmaxf(-CLIPV, hn));
}

// ---------------- DNC memory step -------------------------------------------
// One workgroup (512 threads) per batch element.

__device__ __forceinline__ float block_reduce(float v, float* s, int tid, bool ismax) {
    s[tid] = v;
    __syncthreads();
    for (int st = 256; st > 0; st >>= 1) {
        if (tid < st) s[tid] = ismax ? fmaxf(s[tid], s[tid + st]) : (s[tid] + s[tid + st]);
        __syncthreads();
    }
    float r = s[0];
    __syncthreads();
    return r;
}

__global__ void __launch_bounds__(512)
dnc_step_kernel(const float* __restrict__ xi,     // (B,96) raw interface
                const float* __restrict__ outc,   // (B,512) clipped h1
                float* __restrict__ mem,          // (B,500,16)
                float* __restrict__ link,         // (B,500,500)
                float* __restrict__ prec,         // (B,500)
                float* __restrict__ rw,           // (B,2,500)
                float* __restrict__ wwv,          // (B,500)
                float* __restrict__ usage,        // (B,500)
                float* __restrict__ y)            // (B,544) row block for this t
{
    __shared__ float s_key[512];
    __shared__ int   s_idx[512];
    __shared__ float s_usage[512], s_alloc[512], s_wcw[512], s_ww[512], s_prec[512];
    __shared__ float s_rwold[2][512], s_cw[2][512], s_fwd[2][512], s_bwd[2][512];
    __shared__ float s_red[512];
    __shared__ float s_xi[96];
    __shared__ float s_rk[2][16], s_wk[16], s_ev[16], s_wv[16];
    __shared__ float s_sc[16]; // 0:rs0 1:rs1 2:ws 3:fg0 4:fg1 5:ag 6:wg 7..12:modes

    int b   = blockIdx.x;
    int tid = threadIdx.x;

    float* memb  = mem   + (size_t)b * MCELLS * WCELL;
    float* linkb = link  + (size_t)b * MCELLS * MCELLS;
    float* precb = prec  + (size_t)b * MCELLS;
    float* rwb   = rw    + (size_t)b * RHEADS * MCELLS;
    float* wwb   = wwv   + (size_t)b * MCELLS;
    float* useb  = usage + (size_t)b * MCELLS;

    if (tid < 96) s_xi[tid] = (tid < 93) ? xi[b * 96 + tid] : 0.0f;
    __syncthreads();

    // ---- parse interface vector ----
    if (tid < 32)       s_rk[tid >> 4][tid & 15] = tanhf(s_xi[tid]);
    else if (tid < 48)  s_wk[tid - 32] = tanhf(s_xi[34 + (tid - 32)]);
    else if (tid < 64)  s_ev[tid - 48] = sigf(s_xi[51 + (tid - 48)]);
    else if (tid < 80)  s_wv[tid - 64] = tanhf(s_xi[67 + (tid - 64)]);
    else if (tid == 80) {
        s_sc[0] = softplusf_(s_xi[32]);
        s_sc[1] = softplusf_(s_xi[33]);
        s_sc[2] = softplusf_(s_xi[50]);
        s_sc[3] = sigf(s_xi[83]);
        s_sc[4] = sigf(s_xi[84]);
        s_sc[5] = sigf(s_xi[85]);
        s_sc[6] = sigf(s_xi[86]);
        for (int r = 0; r < 2; ++r) {
            float a = s_xi[87 + 3 * r], bb = s_xi[88 + 3 * r], cc = s_xi[89 + 3 * r];
            float mx = fmaxf(a, fmaxf(bb, cc));
            float ea = __expf(a - mx), eb = __expf(bb - mx), ec = __expf(cc - mx);
            float sd = ea + eb + ec;
            s_sc[7 + 3 * r] = ea / sd;
            s_sc[8 + 3 * r] = eb / sd;
            s_sc[9 + 3 * r] = ec / sd;
        }
    }
    __syncthreads();

    // ---- phase A: retention + usage update, load old state into LDS ----
    if (tid < MCELLS) {
        float r0 = rwb[tid], r1 = rwb[MCELLS + tid];
        s_rwold[0][tid] = r0;
        s_rwold[1][tid] = r1;
        s_prec[tid] = precb[tid];
        float u = useb[tid];
        float w = wwb[tid];
        u = u + (1.0f - u) * w;
        float psi = (1.0f - s_sc[3] * r0) * (1.0f - s_sc[4] * r1);
        s_usage[tid] = u * psi;
    } else {
        s_rwold[0][tid] = 0.0f;
        s_rwold[1][tid] = 0.0f;
        s_prec[tid] = 0.0f;
        s_usage[tid] = 0.0f;
    }
    __syncthreads();

    // ---- phase B: write-key content weights on pre-update memory ----
    float sim = -3.0e38f;
    if (tid < MCELLS) {
        const float* mr = memb + tid * WCELL;
        float nm = 0.0f, dot = 0.0f, kn = 0.0f;
#pragma unroll
        for (int w = 0; w < WCELL; ++w) {
            float mv = mr[w], kv = s_wk[w];
            nm += mv * mv; kn += kv * kv; dot += kv * mv;
        }
        sim = dot / ((sqrtf(nm) + DELTA) * (sqrtf(kn) + DELTA)) * s_sc[2];
    }
    {
        float mx = block_reduce(sim, s_red, tid, true);
        float e  = (tid < MCELLS) ? __expf(sim - mx) : 0.0f;
        float tot = block_reduce(e, s_red, tid, false);
        if (tid < MCELLS) s_wcw[tid] = e / tot;
    }
    __syncthreads();

    // ---- phase C: allocation weights (sort usage ascending) ----
    s_key[tid] = (tid < MCELLS) ? (DELTA + (1.0f - DELTA) * s_usage[tid]) : 2.0f;
    s_idx[tid] = tid;
    __syncthreads();
    for (int k = 2; k <= 512; k <<= 1) {
        for (int j = k >> 1; j > 0; j >>= 1) {
            int ixj = tid ^ j;
            if (ixj > tid) {
                bool up = ((tid & k) == 0);
                float a = s_key[tid], c = s_key[ixj];
                if ((a > c) == up) {
                    s_key[tid] = c; s_key[ixj] = a;
                    int t2 = s_idx[tid]; s_idx[tid] = s_idx[ixj]; s_idx[ixj] = t2;
                }
            }
            __syncthreads();
        }
    }
    if (tid == 0) {
        float excl = 1.0f;
        for (int i = 0; i < MCELLS; ++i) {
            float su = s_key[i];
            s_alloc[s_idx[i]] = (1.0f - su) * excl;
            excl *= su;
        }
    }
    __syncthreads();

    // ---- phase D: write weights ----
    if (tid < MCELLS) {
        float wn = s_sc[6] * (s_sc[5] * s_alloc[tid] + (1.0f - s_sc[5]) * s_wcw[tid]);
        s_ww[tid] = wn;
        wwb[tid] = wn;
        useb[tid] = s_usage[tid];
    } else {
        s_ww[tid] = 0.0f;
    }
    __syncthreads();

    // ---- phase E: memory write ----
    for (int e = tid; e < MCELLS * WCELL; e += 512) {
        int m = e >> 4, w = e & 15;
        float wwm = s_ww[m];
        memb[e] = memb[e] * (1.0f - wwm * s_ev[w]) + wwm * s_wv[w];
    }

    // ---- phase F: temporal link + precedence ----
    float sumww = block_reduce(s_ww[tid], s_red, tid, false);
    for (int e = tid; e < MCELLS * MCELLS; e += 512) {
        int i = e / MCELLS, j = e % MCELLS;
        float l = linkb[e];
        linkb[e] = (i == j) ? 0.0f
                            : (1.0f - s_ww[i] - s_ww[j]) * l + s_ww[i] * s_prec[j];
    }
    if (tid < MCELLS) precb[tid] = (1.0f - sumww) * s_prec[tid] + s_ww[tid];
    __threadfence_block();
    __syncthreads();

    // ---- phase G: read-key content weights on post-update memory ----
    for (int r = 0; r < RHEADS; ++r) {
        float sr = -3.0e38f;
        if (tid < MCELLS) {
            const float* mr = memb + tid * WCELL;
            float nm = 0.0f, dot = 0.0f, kn = 0.0f;
#pragma unroll
            for (int w = 0; w < WCELL; ++w) {
                float mv = mr[w], kv = s_rk[r][w];
                nm += mv * mv; kn += kv * kv; dot += kv * mv;
            }
            sr = dot / ((sqrtf(nm) + DELTA) * (sqrtf(kn) + DELTA)) * s_sc[r];
        }
        float mx = block_reduce(sr, s_red, tid, true);
        float e  = (tid < MCELLS) ? __expf(sr - mx) : 0.0f;
        float tot = block_reduce(e, s_red, tid, false);
        if (tid < MCELLS) s_cw[r][tid] = e / tot;
    }
    __syncthreads();

    // ---- phase H: forward / backward weights via link matvec ----
    for (int r = 0; r < RHEADS; ++r) {
        if (tid < MCELLS) {
            const float* lrow = linkb + (size_t)tid * MCELLS;
            float acc = 0.0f;
            for (int j = 0; j < MCELLS; ++j) acc += lrow[j] * s_rwold[r][j];
            s_fwd[r][tid] = acc;
            float accb = 0.0f;
            for (int i = 0; i < MCELLS; ++i) accb += s_rwold[r][i] * linkb[(size_t)i * MCELLS + tid];
            s_bwd[r][tid] = accb;
        }
    }
    __syncthreads();

    // ---- phase I: new read weights ----
    if (tid < MCELLS) {
        for (int r = 0; r < RHEADS; ++r) {
            float v = s_sc[7 + 3 * r] * s_bwd[r][tid]
                    + s_sc[8 + 3 * r] * s_fwd[r][tid]
                    + s_sc[9 + 3 * r] * s_cw[r][tid];
            s_cw[r][tid] = v;              // reuse as rw_new
            rwb[r * MCELLS + tid] = v;
        }
    }
    __syncthreads();

    // ---- phase J: read vectors + output row ----
    const float* ob = outc + (size_t)b * NHID;
    float* yrow = y + (size_t)b * YDIM;
    if (tid < NHID) yrow[tid] = ob[tid];
    if (tid < RHEADS * WCELL) {
        int r = tid >> 4, w = tid & 15;
        float acc = 0.0f;
        for (int m = 0; m < MCELLS; ++m) acc += s_cw[r][m] * memb[m * WCELL + w];
        yrow[NHID + tid] = acc;
    }
}

// ---------------- host launch -----------------------------------------------

static inline void fill(float* p, float v, int n, hipStream_t s) {
    fill_kernel<<<(n + 255) / 256, 256, 0, s>>>(p, v, n);
}

extern "C" void kernel_launch(void* const* d_in, const int* in_sizes, int n_in,
                              void* d_out, int out_size, void* d_ws, size_t ws_size,
                              hipStream_t stream) {
    const int*   tokens = (const int*)d_in[0];
    const float* emb_w  = (const float*)d_in[1];
    const float* Wih0   = (const float*)d_in[2];
    const float* Whh0   = (const float*)d_in[3];
    const float* bih0   = (const float*)d_in[4];
    const float* bhh0   = (const float*)d_in[5];
    const float* Wih1   = (const float*)d_in[6];
    const float* Whh1   = (const float*)d_in[7];
    const float* bih1   = (const float*)d_in[8];
    const float* bhh1   = (const float*)d_in[9];
    const float* Wif    = (const float*)d_in[10];
    const float* bif    = (const float*)d_in[11];
    const float* Wout   = (const float*)d_in[12];
    const float* bout   = (const float*)d_in[13];
    const float* Wdec   = (const float*)d_in[14];
    const float* bdec   = (const float*)d_in[15];
    float* out = (float*)d_out;

    float* ws = (float*)d_ws;
    size_t off = 0;
    auto alloc = [&](size_t n) { float* p = ws + off; off += n; return p; };

    float* xbuf   = alloc((size_t)T_LEN * BATCH * XDIM);
    float* h0     = alloc(BATCH * NHID);
    float* c0     = alloc(BATCH * NHID);
    float* h1     = alloc(BATCH * NHID);
    float* c1     = alloc(BATCH * NHID);
    float* gates0 = alloc(BATCH * 4 * NHID);
    float* gates1 = alloc(BATCH * 4 * NHID);
    float* outc   = alloc(BATCH * NHID);
    float* xiB    = alloc(BATCH * 96);
    float* memB   = alloc((size_t)BATCH * MCELLS * WCELL);
    float* linkB  = alloc((size_t)BATCH * MCELLS * MCELLS);
    float* precB  = alloc(BATCH * MCELLS);
    float* rwB    = alloc(BATCH * RHEADS * MCELLS);
    float* wwB    = alloc(BATCH * MCELLS);
    float* useB   = alloc(BATCH * MCELLS);
    float* ysB    = alloc((size_t)T_LEN * BATCH * YDIM);
    float* dncB   = alloc((size_t)T_LEN * BATCH * NINP);
    (void)ws_size; (void)in_sizes; (void)n_in; (void)out_size;

    // initial state (re-initialize every call; harness does not re-poison)
    fill(h0, 0.0f, BATCH * NHID, stream);
    fill(c0, 0.0f, BATCH * NHID, stream);
    fill(h1, 0.0f, BATCH * NHID, stream);
    fill(c1, 0.0f, BATCH * NHID, stream);
    fill(memB, DELTA, BATCH * MCELLS * WCELL, stream);
    fill(linkB, 0.0f, BATCH * MCELLS * MCELLS, stream);
    fill(precB, 0.0f, BATCH * MCELLS, stream);
    fill(rwB, DELTA, BATCH * RHEADS * MCELLS, stream);
    fill(wwB, DELTA, BATCH * MCELLS, stream);
    fill(useB, 0.0f, BATCH * MCELLS, stream);

    {
        int total = T_LEN * BATCH * XDIM;
        embed_kernel<<<(total + 255) / 256, 256, 0, stream>>>(tokens, emb_w, xbuf);
    }

    const int H4 = 4 * NHID; // 2048
    dim3 blk(128);
    for (int t = 0; t < T_LEN; ++t) {
        const float* xt = xbuf + (size_t)t * BATCH * XDIM;
        // layer-0 gates: [x_t | h0] @ [Wih0 | Whh0]^T + bih0 + bhh0
        wmma_gemm2<<<dim3((H4 + 63) / 64, 1), blk, 0, stream>>>(
            xt, XDIM, XDIM, h0, NHID, NHID,
            Wih0, Whh0, H4, bih0, bhh0, H4,
            gates0, H4, BATCH, H4);
        lstm_pw_kernel<<<(BATCH * NHID + 255) / 256, 256, 0, stream>>>(gates0, H4, h0, c0, nullptr);
        // layer-1 gates: [h0 | h1] @ [Wih1 | Whh1]^T + bih1 + bhh1
        wmma_gemm2<<<dim3((H4 + 63) / 64, 1), blk, 0, stream>>>(
            h0, NHID, NHID, h1, NHID, NHID,
            Wih1, Whh1, H4, bih1, bhh1, H4,
            gates1, H4, BATCH, H4);
        lstm_pw_kernel<<<(BATCH * NHID + 255) / 256, 256, 0, stream>>>(gates1, H4, h1, c1, outc);
        // interface vector (N padded to 96, W rows >= 93 read as 0)
        wmma_gemm2<<<dim3((96 + 63) / 64, 1), blk, 0, stream>>>(
            outc, NHID, NHID, nullptr, 0, 0,
            Wif, nullptr, 93, bif, nullptr, 93,
            xiB, 96, BATCH, 96);
        // DNC memory step (one WG per batch element)
        dnc_step_kernel<<<BATCH, 512, 0, stream>>>(
            xiB, outc, memB, linkB, precB, rwB, wwB, useB,
            ysB + (size_t)t * BATCH * YDIM);
    }

    const int MT = T_LEN * BATCH; // 700
    // dnc_out = ys @ Wout^T + bout
    wmma_gemm2<<<dim3((NINP + 63) / 64, (MT + 31) / 32), blk, 0, stream>>>(
        ysB, YDIM, YDIM, nullptr, 0, 0,
        Wout, nullptr, NINP, bout, nullptr, NINP,
        dncB, NINP, MT, NINP);
    // logits = dnc_out @ Wdec^T + bdec
    wmma_gemm2<<<dim3((NTOKEN + 63) / 64, (MT + 31) / 32), blk, 0, stream>>>(
        dncB, NINP, NINP, nullptr, 0, 0,
        Wdec, nullptr, NTOKEN, bdec, nullptr, NTOKEN,
        out, NTOKEN, MT, NTOKEN);
}
